// MultiHead_LayerReduce_66700842107356
// MI455X (gfx1250) — matrix operand order repaired and checked by reading, not compiled, not run on hardware
//
#include <hip/hip_runtime.h>
#include <hip/hip_bf16.h>

typedef __attribute__((ext_vector_type(16))) __bf16 v16bf;
typedef __attribute__((ext_vector_type(8)))  __bf16 v8bf;
typedef __attribute__((ext_vector_type(2)))  __bf16 v2bf;
typedef __attribute__((ext_vector_type(8)))  float  v8f;
typedef __attribute__((ext_vector_type(2)))  unsigned int v2u;
typedef __attribute__((ext_vector_type(4)))  unsigned int v4u;
typedef __attribute__((ext_vector_type(8)))  unsigned int v8u;

// Shapes fixed by the reference: B=8, C=256, N=1024, H=8, HB=64, C8=32, DO=256

__device__ __forceinline__ unsigned int pack_bf16(float x, float y)
{
    v2bf p;
    p[0] = (__bf16)x;
    p[1] = (__bf16)y;                 // lowers to v_cvt_pk_bf16_f32
    return __builtin_bit_cast(unsigned int, p);
}

// ---------------------------------------------------------------------------
// Generic batched GEMM:  D[g] = A[g?] (MxK) @ Bm[g] (KxNn)  (+bias[m])
// f32 operands staged to LDS as packed bf16 K-pairs (the exact per-VGPR WMMA
// operand format), accumulate f32 via v_wmma_f32_16x16x32_bf16.
// WAVES_M*2 waves; block = (WAVES_M*32) x 64; 2x2 16x16 WMMA tiles per wave.
// Double-buffered LDS staging: one barrier per K-step, global loads for the
// next tile issue before the current tile's WMMAs.
// K % 32 == 0, Nn % 64 == 0.  M arbitrary (guarded; hoisted for full tiles).
// ---------------------------------------------------------------------------
template<bool TRANS_OUT, bool BATCH_A, bool HAS_BIAS, int WAVES_M>
__global__ __launch_bounds__(WAVES_M * 64)
void gemm_wmma(const float* __restrict__ A, const float* __restrict__ Bm,
               const float* __restrict__ bias, float* __restrict__ D,
               int M, int K, int Nn, long sA, long sB, long sD)
{
    constexpr int BM      = WAVES_M * 32;
    constexpr int THREADS = WAVES_M * 64;
    constexpr int AITER   = 4;                    // BM*8 / THREADS == 4 always
    constexpr int BITER   = 256 / THREADS;
    // a[m][k/2]: pitch 20 dwords -> rows 16B aligned, conflict-free b128 reads
    // b[k/2][n]: pitch 68 dwords -> rows 16B aligned, half-waves bank-disjoint
    constexpr int TILE    = BM * 20 + 16 * 68;    // dwords per buffer
    constexpr int SMEM_DW = (2 * TILE > BM * 65) ? 2 * TILE
                            : (TRANS_OUT ? BM * 65 : 2 * TILE);

    const int g  = blockIdx.z;
    const int bm = blockIdx.y * BM;
    const int bn = blockIdx.x << 6;
    const float* Ag = BATCH_A ? (A + (long)g * sA) : A;
    const float* Bg = Bm + (long)g * sB;
    float*       Dg = D  + (long)g * sD;

    __shared__ unsigned int smem[SMEM_DW];

    const int tid  = threadIdx.x;
    const int lane = tid & 31;
    const int wave = tid >> 5;
    const int mo = (wave >> 1) << 5;   // 0,32,..,BM-32
    const int no = (wave & 1)  << 5;   // 0 / 32

    v8f acc[2][2];
#pragma unroll
    for (int i = 0; i < 2; ++i)
#pragma unroll
        for (int j = 0; j < 2; ++j)
#pragma unroll
            for (int e = 0; e < 8; ++e) acc[i][j][e] = 0.0f;

    // CDNA5 WMMA wave32 operand indices
    const int am  = lane & 15;              // A: M row within 16
    const int akp = (lane >> 4) << 2;       // A: packed-K col base (0 / 4)
    const int bnl = lane & 15;              // B: N col within 16
    const int bkp = (lane >> 4) << 2;       // B: packed-K row base (0 / 4)

    const bool fullA = (bm + BM <= M);

    // loop-invariant staging coordinates
    int rA[AITER], cA[AITER];
#pragma unroll
    for (int i = 0; i < AITER; ++i) {
        int e = i * THREADS + tid;          // BM*8 float4 tasks over BMx32 A tile
        rA[i] = e >> 3;
        cA[i] = (e & 7) << 2;
    }
    int kpB[BITER], n0B[BITER];
#pragma unroll
    for (int i = 0; i < BITER; ++i) {
        int task = i * THREADS + tid;       // 256 pair-pack tasks over 32x64 B tile
        kpB[i] = task >> 4;
        n0B[i] = (task & 15) << 2;
    }

    auto stage = [&](int buf, int k0) {
        unsigned int* aB = smem + buf * TILE;
        unsigned int* bB = aB + BM * 20;
#pragma unroll
        for (int i = 0; i < AITER; ++i) {
            float4 ld = make_float4(0.f, 0.f, 0.f, 0.f);
            if (fullA || bm + rA[i] < M)
                ld = *(const float4*)(Ag + (long)(bm + rA[i]) * K + k0 + cA[i]);
            v2u p;
            p[0] = pack_bf16(ld.x, ld.y);
            p[1] = pack_bf16(ld.z, ld.w);
            *(v2u*)&aB[rA[i] * 20 + (cA[i] >> 1)] = p;
        }
#pragma unroll
        for (int i = 0; i < BITER; ++i) {
            const float* src = Bg + (long)(k0 + 2 * kpB[i]) * Nn + bn + n0B[i];
            float4 r0 = *(const float4*)(src);
            float4 r1 = *(const float4*)(src + Nn);
            v4u p;
            p[0] = pack_bf16(r0.x, r1.x);
            p[1] = pack_bf16(r0.y, r1.y);
            p[2] = pack_bf16(r0.z, r1.z);
            p[3] = pack_bf16(r0.w, r1.w);
            *(v4u*)&bB[kpB[i] * 68 + n0B[i]] = p;
        }
    };

    auto compute = [&](int buf) {
        const unsigned int* aB = smem + buf * TILE;
        const unsigned int* bB = aB + BM * 20;
        v16bf af[2], bfr[2];
#pragma unroll
        for (int t2 = 0; t2 < 2; ++t2) {
            int row = mo + t2 * 16 + am;
            v4u lo = *(const v4u*)&aB[row * 20 + akp];
            v4u hi = *(const v4u*)&aB[row * 20 + 8 + akp];
            v8bf l = __builtin_bit_cast(v8bf, lo);
            v8bf h = __builtin_bit_cast(v8bf, hi);
            af[t2] = __builtin_shufflevector(
                l, h, 0, 1, 2, 3, 4, 5, 6, 7, 8, 9, 10, 11, 12, 13, 14, 15);

            int coln = no + t2 * 16 + bnl;
            v8u tb;
#pragma unroll
            for (int v = 0; v < 8; ++v) tb[v] = bB[(bkp + v + (v & 4)) * 68 + coln];
            bfr[t2] = __builtin_bit_cast(v16bf, tb);
        }
#pragma unroll
        for (int mi = 0; mi < 2; ++mi)
#pragma unroll
            for (int ni = 0; ni < 2; ++ni)
                acc[mi][ni] = __builtin_amdgcn_wmma_f32_16x16x32_bf16(
                    false, af[mi], false, bfr[ni], (short)0, acc[mi][ni],
                    false, false);
    };

    // software pipeline: stage(next) -> compute(cur) -> barrier
    stage(0, 0);
    __syncthreads();
    int p = 0;
    for (int k0 = 0; k0 < K; k0 += 32) {
        if (k0 + 32 < K) stage(p ^ 1, k0 + 32);
        compute(p);
        __syncthreads();
        p ^= 1;
    }

    const int dn  = lane & 15;
    const int dmb = (lane >> 4) << 3;
    if constexpr (!TRANS_OUT) {
#pragma unroll
        for (int mi = 0; mi < 2; ++mi)
#pragma unroll
            for (int ni = 0; ni < 2; ++ni) {
                int col = bn + no + ni * 16 + dn;
#pragma unroll
                for (int v = 0; v < 8; ++v) {
                    int row = bm + mo + mi * 16 + dmb + v;
                    if (row < M) {
                        float val = acc[mi][ni][v];
                        if constexpr (HAS_BIAS) val += bias[row];
                        Dg[(long)row * Nn + col] = val;
                    }
                }
            }
    } else {
        // stage in LDS, then store transposed ([Nn, M]) with coalesced runs in M
        float* dsm = (float*)smem;          // tiles are dead after the k-loop
#pragma unroll
        for (int mi = 0; mi < 2; ++mi)
#pragma unroll
            for (int ni = 0; ni < 2; ++ni)
#pragma unroll
                for (int v = 0; v < 8; ++v)
                    dsm[(mo + mi * 16 + dmb + v) * 65 + no + ni * 16 + dn] =
                        acc[mi][ni][v];
        __syncthreads();
        for (int e = tid; e < BM * 64; e += THREADS) {
            int cc = e >> (WAVES_M == 2 ? 6 : 7);
            int r  = e & (BM - 1);
            int row = bm + r;
            if (row < M) {
                float val = dsm[r * 65 + cc];
                if constexpr (HAS_BIAS) val += bias[row];
                Dg[(long)(bn + cc) * M + row] = val;
            }
        }
    }
}

// ---------------------------------------------------------------------------
// split-heads:  qbar[h*8+b][cp][np] = qf[b][4*cp + (np>>8)][h*256 + (np&255)]
// qf: [8][1024][2048] row-major, qbar: [64][256][1024]
// ---------------------------------------------------------------------------
__global__ __launch_bounds__(256)
void split_heads_kernel(const float* __restrict__ qf, float* __restrict__ qbar)
{
    long i = (long)blockIdx.x * 256 + threadIdx.x;      // 0 .. 16M-1
    int np = (int)(i & 1023);
    int cp = (int)((i >> 10) & 255);
    int g  = (int)(i >> 18);
    int h = g >> 3, b = g & 7;
    int n = cp * 4 + (np >> 8);
    int c = np & 255;
    qbar[i] = qf[((long)b * 1024 + n) * 2048 + h * 256 + c];
}

// xkcat[g][32+j][n] = kbar[g][index[j]][n]
__global__ __launch_bounds__(256)
void gather_concat_kernel(const float* __restrict__ kbar,
                          const int* __restrict__ index,
                          float* __restrict__ xkcat)
{
    long i = (long)blockIdx.x * 256 + threadIdx.x;      // 0 .. 2M-1
    int n = (int)(i & 1023);
    int j = (int)((i >> 10) & 31);
    int g = (int)(i >> 15);
    xkcat[((long)g * 64 + 32 + j) * 1024 + n] =
        kbar[((long)g * 256 + index[j]) * 1024 + n];
}

// softmax over 32 channels:  xk [64][32][1024], softmax along dim 1
__global__ __launch_bounds__(256)
void softmax_ch32_kernel(float* __restrict__ xk)
{
    long i = (long)blockIdx.x * 256 + threadIdx.x;      // 0 .. 65535
    int n = (int)(i & 1023);
    int g = (int)(i >> 10);
    float* p = xk + (long)g * 32 * 1024 + n;
    float mx = -3.4e38f;
    for (int c = 0; c < 32; ++c) mx = fmaxf(mx, p[(long)c * 1024]);
    float s = 0.f;
    for (int c = 0; c < 32; ++c) s += expf(p[(long)c * 1024] - mx);
    float inv = 1.f / s;
    for (int c = 0; c < 32; ++c) {
        long o = (long)c * 1024;
        p[o] = expf(p[o] - mx) * inv;
    }
}

// row softmax over 1024 elements; one block per row
__global__ __launch_bounds__(256)
void softmax_row1024_kernel(float* __restrict__ attn)
{
    float* p = attn + (long)blockIdx.x * 1024;
    int t = threadIdx.x;
    float v[4];
    float mx = -3.4e38f;
#pragma unroll
    for (int i = 0; i < 4; ++i) { v[i] = p[t + i * 256]; mx = fmaxf(mx, v[i]); }
    __shared__ float red[256];
    red[t] = mx; __syncthreads();
    for (int s = 128; s > 0; s >>= 1) {
        if (t < s) red[t] = fmaxf(red[t], red[t + s]);
        __syncthreads();
    }
    mx = red[0]; __syncthreads();
    float sum = 0.f;
#pragma unroll
    for (int i = 0; i < 4; ++i) { v[i] = expf(v[i] - mx); sum += v[i]; }
    red[t] = sum; __syncthreads();
    for (int s = 128; s > 0; s >>= 1) {
        if (t < s) red[t] += red[t + s];
        __syncthreads();
    }
    float inv = 1.f / red[0];
#pragma unroll
    for (int i = 0; i < 4; ++i) p[t + i * 256] = v[i] * inv;
}

// colsum[g][n] = sum_q attn[g][q][n]
__global__ __launch_bounds__(256)
void colsum_kernel(const float* __restrict__ attn, float* __restrict__ colsum)
{
    long i = (long)blockIdx.x * 256 + threadIdx.x;      // 0 .. 65535
    int n = (int)(i & 1023);
    int g = (int)(i >> 10);
    const float* p = attn + (long)g * 1024 * 1024 + n;
    float s = 0.f;
    for (int q = 0; q < 1024; ++q) s += p[(long)q * 1024];
    colsum[i] = s;
}

__global__ __launch_bounds__(256)
void renorm_kernel(float* __restrict__ attn, const float* __restrict__ colsum)
{
    long i = (long)blockIdx.x * 256 + threadIdx.x;      // 0 .. 64M-1
    int n = (int)(i & 1023);
    int g = (int)(i >> 20);
    attn[i] = attn[i] / (1e-9f + colsum[(long)g * 1024 + n]);
}

// BN stats per channel: x [64][256][1024]; mean over dims (0,2)
__global__ __launch_bounds__(256)
void bn_stats_kernel(const float* __restrict__ x, float* __restrict__ mr)
{
    int c = blockIdx.x, t = threadIdx.x;
    float s = 0.f, s2 = 0.f;
    for (int g = 0; g < 64; ++g) {
        const float* p = x + (long)g * 262144 + (long)c * 1024;
        for (int n = t; n < 1024; n += 256) {
            float v = p[n];
            s += v; s2 += v * v;
        }
    }
    __shared__ float rs[256], rq[256];
    rs[t] = s; rq[t] = s2; __syncthreads();
    for (int k = 128; k > 0; k >>= 1) {
        if (t < k) { rs[t] += rs[t + k]; rq[t] += rq[t + k]; }
        __syncthreads();
    }
    if (t == 0) {
        float mu  = rs[0] * (1.f / 65536.f);
        float var = rq[0] * (1.f / 65536.f) - mu * mu;
        mr[c]       = mu;
        mr[256 + c] = rsqrtf(var + 1e-5f);
    }
}

__global__ __launch_bounds__(256)
void bn_apply_kernel(float* __restrict__ x, const float* __restrict__ mr,
                     const float* __restrict__ gw, const float* __restrict__ gb,
                     const float* __restrict__ alpha, const float* __restrict__ beta)
{
    long i = (long)blockIdx.x * 256 + threadIdx.x;      // 0 .. 16M-1
    int c = (int)((i >> 10) & 255);
    float v = gw[c] * (x[i] - mr[c]) * mr[256 + c] + gb[c];
    v = fmaxf(v, 0.f);
    x[i] = alpha[c] * v + beta[c];
}

// merge-heads: pre[b][o][n2] = xo[(o>>8)*8+b][n2>>2][((n2&3)<<8)|(o&255)]
__global__ __launch_bounds__(256)
void merge_heads_kernel(const float* __restrict__ xo, float* __restrict__ pre)
{
    long i = (long)blockIdx.x * 256 + threadIdx.x;      // 0 .. 16M-1
    int n2 = (int)(i & 1023);
    int o  = (int)((i >> 10) & 2047);
    int b  = (int)(i >> 21);
    int h = o >> 8, c2 = o & 255;
    int g = h * 8 + b;
    int c = n2 >> 2;
    int n = ((n2 & 3) << 8) | c2;
    pre[i] = xo[((long)g * 256 + c) * 1024 + n];
}

// ---------------------------------------------------------------------------
extern "C" void kernel_launch(void* const* d_in, const int* in_sizes, int n_in,
                              void* d_out, int out_size, void* d_ws, size_t ws_size,
                              hipStream_t stream)
{
    const float* x     = (const float*)d_in[0];
    const int*   index = (const int*)  d_in[1];
    const float* Wq    = (const float*)d_in[2];
    const float* Wk    = (const float*)d_in[3];
    const float* Wv    = (const float*)d_in[4];
    const float* bv    = (const float*)d_in[5];
    const float* fcq_w = (const float*)d_in[6];
    const float* fcq_b = (const float*)d_in[7];
    const float* fck_w = (const float*)d_in[8];
    const float* fck_b = (const float*)d_in[9];
    const float* fcv_w = (const float*)d_in[10];
    const float* fcv_b = (const float*)d_in[11];
    const float* lr_qw = (const float*)d_in[12];
    const float* lr_kw = (const float*)d_in[13];
    const float* lr_k2c= (const float*)d_in[14];
    const float* lr_vw = (const float*)d_in[15];
    const float* lr_vb = (const float*)d_in[16];
    const float* lr_tw = (const float*)d_in[17];
    const float* lr_tb = (const float*)d_in[18];
    const float* bn_g  = (const float*)d_in[19];
    const float* bn_b  = (const float*)d_in[20];
    const float* alpha = (const float*)d_in[21];
    const float* beta  = (const float*)d_in[22];
    const float* fco_w = (const float*)d_in[23];
    const float* fco_b = (const float*)d_in[24];

    float* attn = (float*)d_out;                            // [64,1024,1024]
    float* outF = (float*)d_out + (size_t)64 * 1024 * 1024; // [8,256,1024]

    float* ws = (float*)d_ws;
    size_t off = 0;
    auto alloc = [&](size_t n) { float* p = ws + off; off += n; return p; };
    float* S0  = alloc(2097152);   // conv out     [8,256,1024]
    float* A1  = alloc(16777216);  // fc out / xv / xr2(xo)
    float* A2  = alloc(16777216);  // split-heads / xr / pre
    float* XQ  = alloc(2097152);   // [64,1024,32]
    float* XKC = alloc(4194304);   // [64,64,1024]
    float* XK  = alloc(2097152);   // [64,32,1024]
    float* CS  = alloc(65536);     // attn column sums
    float* MR  = alloc(512);       // bn mean/rstd
    (void)ws_size; (void)in_sizes; (void)n_in; (void)out_size;

    // ---- Q path ----
    gemm_wmma<false,false,false,4><<<dim3(16,2,8),  256,0,stream>>>(Wq,    x,  nullptr, S0, 256, 256,1024, 0,      262144, 262144);
    gemm_wmma<true, false,true ,4><<<dim3(16,16,8), 256,0,stream>>>(fcq_w, S0, fcq_b,   A1, 2048,256,1024, 0,      262144, 2097152);
    split_heads_kernel<<<65536,256,0,stream>>>(A1, A2);
    gemm_wmma<true, false,false,2><<<dim3(16,1,64), 128,0,stream>>>(lr_qw, A2, nullptr, XQ, 32,  256,1024, 0,      262144, 32768);

    // ---- K path ----
    gemm_wmma<false,false,false,4><<<dim3(16,2,8),  256,0,stream>>>(Wk,    x,  nullptr, S0, 256, 256,1024, 0,      262144, 262144);
    gemm_wmma<true, false,true ,4><<<dim3(16,16,8), 256,0,stream>>>(fck_w, S0, fck_b,   A1, 2048,256,1024, 0,      262144, 2097152);
    split_heads_kernel<<<65536,256,0,stream>>>(A1, A2);
    gemm_wmma<false,false,false,2><<<dim3(16,1,64), 128,0,stream>>>(lr_kw, A2, nullptr, XKC,32,  256,1024, 0,      262144, 65536);
    gather_concat_kernel<<<8192,256,0,stream>>>(A2, index, XKC);
    gemm_wmma<false,false,false,2><<<dim3(16,1,64), 128,0,stream>>>(lr_k2c,XKC,nullptr, XK, 32,  64, 1024, 0,      65536,  32768);
    softmax_ch32_kernel<<<256,256,0,stream>>>(XK);

    // ---- V path ----
    gemm_wmma<false,false,true ,4><<<dim3(16,2,8),  256,0,stream>>>(Wv,    x,  bv,      S0, 256, 256,1024, 0,      262144, 262144);
    gemm_wmma<true, false,true ,4><<<dim3(16,16,8), 256,0,stream>>>(fcv_w, S0, fcv_b,   A1, 2048,256,1024, 0,      262144, 2097152);
    split_heads_kernel<<<65536,256,0,stream>>>(A1, A2);
    gemm_wmma<false,false,true ,4><<<dim3(16,2,64), 256,0,stream>>>(lr_vw, A2, lr_vb,   A1, 256, 256,1024, 0,      262144, 262144);

    // ---- attention core ----
    gemm_wmma<false,true, false,4><<<dim3(16,8,64), 256,0,stream>>>(XQ,   XK,  nullptr, attn,1024,32, 1024, 32768,  32768,  1048576);
    softmax_row1024_kernel<<<65536,256,0,stream>>>(attn);
    colsum_kernel<<<256,256,0,stream>>>(attn, CS);
    renorm_kernel<<<262144,256,0,stream>>>(attn, CS);
    gemm_wmma<false,true, false,4><<<dim3(16,2,64), 256,0,stream>>>(A1,   attn,nullptr, A2, 256, 1024,1024, 262144, 1048576,262144);

    // ---- tail: lr_t, BN+ReLU+affine, merge, fco ----
    gemm_wmma<false,false,true ,4><<<dim3(16,2,64), 256,0,stream>>>(lr_tw, A2, lr_tb,   A1, 256, 256,1024, 0,      262144, 262144);
    bn_stats_kernel<<<256,256,0,stream>>>(A1, MR);
    bn_apply_kernel<<<65536,256,0,stream>>>(A1, MR, bn_g, bn_b, alpha, beta);
    merge_heads_kernel<<<65536,256,0,stream>>>(A1, A2);
    gemm_wmma<false,false,true ,4><<<dim3(16,2,8),  256,0,stream>>>(fco_w, A2, fco_b,   outF,256, 2048,1024, 0,     2097152,262144);
}
